// SAGE_4020089389331
// MI455X (gfx1250) — compile-verified, matrix operand order checked by reading, not run on hardware
//
#include <hip/hip_runtime.h>
#include <hip/hip_bf16.h>

#define N_NODES 100000
#define N_EDGES 800000
#define F_IN    256
#define F_HID   256
#define F_OUT   64

typedef __attribute__((ext_vector_type(16))) __bf16 v16bf;
typedef __attribute__((ext_vector_type(8)))  __bf16 v8bf;
typedef __attribute__((ext_vector_type(8)))  float  v8f;
typedef __attribute__((ext_vector_type(4)))  float  v4f;

// ---------------------------------------------------------------------------
// Load 16 bf16 per lane in the CDNA5 16-bit A/B operand layout:
//   elements 0..7  <- p[0..7]      (K = k0 + half*8 + 0..7)
//   elements 8..15 <- p[16..23]    (K = k0 + 16 + half*8 + 0..7)
// p must point at row + k0 + half*8 (16B aligned) -> two b128 loads.
// ---------------------------------------------------------------------------
__device__ __forceinline__ v16bf load_ab_tile(const __bf16* p) {
    v8bf lo = *(const v8bf*)(p);
    v8bf hi = *(const v8bf*)(p + 16);
    v16bf r;
#pragma unroll
    for (int i = 0; i < 8; ++i) { r[i] = lo[i]; r[i + 8] = hi[i]; }
    return r;
}

// ---------------------------------------------------------------------------
// Degree / inverse degree
// ---------------------------------------------------------------------------
__global__ void sage_deg_kernel(const int* __restrict__ dst, float* __restrict__ deg) {
    int e = blockIdx.x * blockDim.x + threadIdx.x;
    if (e < N_EDGES) atomicAdd(deg + dst[e], 1.0f);
}

__global__ void sage_invdeg_kernel(float* __restrict__ deg) {
    int i = blockIdx.x * blockDim.x + threadIdx.x;
    if (i < N_NODES) deg[i] = 1.0f / fmaxf(deg[i], 1.0f);
}

// ---------------------------------------------------------------------------
// f32 -> bf16 (n divisible by 4)
// ---------------------------------------------------------------------------
__global__ void sage_f32_to_bf16(const float* __restrict__ x, __bf16* __restrict__ y,
                                 long long n) {
    long long i = ((long long)blockIdx.x * blockDim.x + threadIdx.x) * 4;
    if (i < n) {
        v4f v = *(const v4f*)(x + i);
        y[i + 0] = (__bf16)v.x;
        y[i + 1] = (__bf16)v.y;
        y[i + 2] = (__bf16)v.z;
        y[i + 3] = (__bf16)v.w;
    }
}

// ---------------------------------------------------------------------------
// Weight transpose + bf16 convert: W[K][N] row-major -> Wt[N][K] (n-major)
// ---------------------------------------------------------------------------
__global__ void sage_wt_bf16(const float* __restrict__ W, __bf16* __restrict__ Wt,
                             int K, int N) {
    int idx = blockIdx.x * blockDim.x + threadIdx.x;
    if (idx < K * N) {
        int n = idx / K;
        int k = idx - n * K;
        Wt[idx] = (__bf16)W[(size_t)k * N + n];
    }
}

// ---------------------------------------------------------------------------
// Edge scatter: agg[dst] += h[src] * inv_deg[dst]   (64 threads / edge, float4)
// ---------------------------------------------------------------------------
__global__ void sage_aggregate(const float* __restrict__ h,
                               const int* __restrict__ src,
                               const int* __restrict__ dst,
                               const float* __restrict__ invDeg,
                               float* __restrict__ agg) {
    long long g = (long long)blockIdx.x * blockDim.x + threadIdx.x;
    int e = (int)(g >> 6);
    int c = ((int)g & 63) * 4;
    if (e >= N_EDGES) return;
    int s = src[e];
    int d = dst[e];
    float w = invDeg[d];
    v4f v = *(const v4f*)(h + (size_t)s * 256 + c);
    float* o = agg + (size_t)d * 256 + c;
    atomicAdd(o + 0, v.x * w);
    atomicAdd(o + 1, v.y * w);
    atomicAdd(o + 2, v.z * w);
    atomicAdd(o + 3, v.w * w);
}

// ---------------------------------------------------------------------------
// Fused SAGE layer GEMM:
//   out = [relu]( Aself @ WsT^T + Aagg @ WnT^T + bias )
// K = 256 for both operands. Each wave -> 16(M) x 64(N) tile via 4 f32 accums.
// nOut == 256: blockDim 256 (8 waves, 2xM x 4xN), block tile 32 x 256.
// nOut ==  64: blockDim 64  (2 waves, 2xM x 1xN), block tile 32 x 64.
// 32 | 100000, so tiling is exact and EXEC stays all-ones for WMMA.
// ---------------------------------------------------------------------------
__global__ __launch_bounds__(256) void sage_gemm_wmma(
        const __bf16* __restrict__ Aself,
        const __bf16* __restrict__ Aagg,
        const __bf16* __restrict__ WsT,
        const __bf16* __restrict__ WnT,
        const float*  __restrict__ bias,
        float* outF, __bf16* outB,
        int nOut, int doRelu) {
    const int K = 256;
    const int lane  = threadIdx.x & 31;
    const int wave  = threadIdx.x >> 5;
    const int wavesPerN = nOut >> 6;                 // waves tiling N (64 cols each)
    const int waveM = wave / wavesPerN;
    const int waveN = wave - waveM * wavesPerN;
    const int rowsPerBlock = (((int)blockDim.x >> 5) / wavesPerN) * 16;
    const int m0   = blockIdx.x * rowsPerBlock + waveM * 16;
    const int half = (lane >> 4) & 1;
    const int l16  = lane & 15;

    v8f acc[4] = {};

    const __bf16* Abase[2] = { Aself, Aagg };
    const __bf16* Wbase[2] = { WsT,   WnT  };

#pragma unroll
    for (int ph = 0; ph < 2; ++ph) {
        const __bf16* Arow = Abase[ph] + (size_t)(m0 + l16) * K + half * 8;
        const __bf16* Wb   = Wbase[ph] + half * 8;
#pragma unroll
        for (int k0 = 0; k0 < K; k0 += 32) {
            v16bf a = load_ab_tile(Arow + k0);
#pragma unroll
            for (int t = 0; t < 4; ++t) {
                int n = waveN * 64 + t * 16 + l16;
                v16bf b = load_ab_tile(Wb + (size_t)n * K + k0);
                acc[t] = __builtin_amdgcn_wmma_f32_16x16x32_bf16(
                    /*neg_a=*/false, a, /*neg_b=*/false, b,
                    /*c_mod=*/(short)0, acc[t],
                    /*reuse_a=*/false, /*reuse_b=*/false);
            }
        }
    }

    // All reads of (possibly aliased) Aself complete before any stores.
    __syncthreads();

#pragma unroll
    for (int t = 0; t < 4; ++t) {
        int n = waveN * 64 + t * 16 + l16;
        float bv = bias[n];
#pragma unroll
        for (int r = 0; r < 8; ++r) {
            int m = m0 + half * 8 + r;
            float v = acc[t][r] + bv;
            if (doRelu) v = fmaxf(v, 0.0f);
            size_t o = (size_t)m * nOut + n;
            outF[o] = v;
            if (outB) outB[o] = (__bf16)v;
        }
    }
}

// ---------------------------------------------------------------------------
// Host launcher
// ---------------------------------------------------------------------------
extern "C" void kernel_launch(void* const* d_in, const int* in_sizes, int n_in,
                              void* d_out, int out_size, void* d_ws, size_t ws_size,
                              hipStream_t stream) {
    const float* x    = (const float*)d_in[0];
    const int*   src  = (const int*)  d_in[1];
    const int*   dst  = (const int*)  d_in[2];
    const float* Ws1  = (const float*)d_in[3];
    const float* Wn1  = (const float*)d_in[4];
    const float* b1   = (const float*)d_in[5];
    const float* Ws2  = (const float*)d_in[6];
    const float* Wn2  = (const float*)d_in[7];
    const float* b2   = (const float*)d_in[8];
    const float* Ws3  = (const float*)d_in[9];
    const float* Wn3  = (const float*)d_in[10];
    const float* b3   = (const float*)d_in[11];
    float* out = (float*)d_out;

    // --- carve workspace ---------------------------------------------------
    char* ws = (char*)d_ws;
    size_t off = 0;
    auto carve = [&](size_t bytes) -> void* {
        void* p = ws + off;
        off += (bytes + 255) & ~(size_t)255;
        return p;
    };
    float*  invDeg = (float*) carve((size_t)N_NODES * 4);
    __bf16* xb     = (__bf16*)carve((size_t)N_NODES * F_IN * 2);
    float*  hF     = (float*) carve((size_t)N_NODES * 256 * 4);
    __bf16* hB     = (__bf16*)carve((size_t)N_NODES * 256 * 2);
    float*  aggF   = (float*) carve((size_t)N_NODES * 256 * 4);
    __bf16* aggB   = (__bf16*)carve((size_t)N_NODES * 256 * 2);
    __bf16* Ws1T   = (__bf16*)carve((size_t)256 * 256 * 2);
    __bf16* Wn1T   = (__bf16*)carve((size_t)256 * 256 * 2);
    __bf16* Ws2T   = (__bf16*)carve((size_t)256 * 256 * 2);
    __bf16* Wn2T   = (__bf16*)carve((size_t)256 * 256 * 2);
    __bf16* Ws3T   = (__bf16*)carve((size_t)256 * 64 * 2);
    __bf16* Wn3T   = (__bf16*)carve((size_t)256 * 64 * 2);
    (void)ws_size; (void)in_sizes; (void)n_in; (void)out_size;

    const long long feat_elems = (long long)N_NODES * 256;

    // --- degree + inverse degree ------------------------------------------
    hipMemsetAsync(invDeg, 0, (size_t)N_NODES * 4, stream);
    sage_deg_kernel<<<(N_EDGES + 255) / 256, 256, 0, stream>>>(dst, invDeg);
    sage_invdeg_kernel<<<(N_NODES + 255) / 256, 256, 0, stream>>>(invDeg);

    // --- operand conversion ------------------------------------------------
    sage_f32_to_bf16<<<(int)(feat_elems / 4 + 255) / 256, 256, 0, stream>>>(x, xb, feat_elems);
    sage_wt_bf16<<<(256 * 256 + 255) / 256, 256, 0, stream>>>(Ws1, Ws1T, 256, 256);
    sage_wt_bf16<<<(256 * 256 + 255) / 256, 256, 0, stream>>>(Wn1, Wn1T, 256, 256);
    sage_wt_bf16<<<(256 * 256 + 255) / 256, 256, 0, stream>>>(Ws2, Ws2T, 256, 256);
    sage_wt_bf16<<<(256 * 256 + 255) / 256, 256, 0, stream>>>(Wn2, Wn2T, 256, 256);
    sage_wt_bf16<<<(256 * 64 + 255) / 256, 256, 0, stream>>>(Ws3, Ws3T, 256, 64);
    sage_wt_bf16<<<(256 * 64 + 255) / 256, 256, 0, stream>>>(Wn3, Wn3T, 256, 64);

    const int aggBlocks  = (int)(((long long)N_EDGES * 64 + 255) / 256);
    const int convBlocks = (int)(feat_elems / 4 / 256);
    const int gemmBlocks = N_NODES / 32;   // 32 | 100000 -> exact

    // --- layer 1 -----------------------------------------------------------
    hipMemsetAsync(aggF, 0, (size_t)N_NODES * 256 * 4, stream);
    sage_aggregate<<<aggBlocks, 256, 0, stream>>>(x, src, dst, invDeg, aggF);
    sage_f32_to_bf16<<<convBlocks, 256, 0, stream>>>(aggF, aggB, feat_elems);
    sage_gemm_wmma<<<gemmBlocks, 256, 0, stream>>>(xb, aggB, Ws1T, Wn1T, b1,
                                                   hF, hB, 256, 1);

    // --- layer 2 (in-place on hF/hB; __syncthreads orders reads vs writes) --
    hipMemsetAsync(aggF, 0, (size_t)N_NODES * 256 * 4, stream);
    sage_aggregate<<<aggBlocks, 256, 0, stream>>>(hF, src, dst, invDeg, aggF);
    sage_f32_to_bf16<<<convBlocks, 256, 0, stream>>>(aggF, aggB, feat_elems);
    sage_gemm_wmma<<<gemmBlocks, 256, 0, stream>>>(hB, aggB, Ws2T, Wn2T, b2,
                                                   hF, hB, 256, 1);

    // --- layer 3 (no relu, f32 out to d_out) -------------------------------
    hipMemsetAsync(aggF, 0, (size_t)N_NODES * 256 * 4, stream);
    sage_aggregate<<<aggBlocks, 256, 0, stream>>>(hF, src, dst, invDeg, aggF);
    sage_f32_to_bf16<<<convBlocks, 256, 0, stream>>>(aggF, aggB, feat_elems);
    sage_gemm_wmma<<<gemmBlocks, 64, 0, stream>>>(hB, aggB, Ws3T, Wn3T, b3,
                                                  out, (__bf16*)nullptr, 64, 0);
}
